// QuantVoxelBackBone8x_15899968930297
// MI455X (gfx1250) — matrix operand order repaired and checked by reading, not compiled, not run on hardware
//
#include <hip/hip_runtime.h>
#include <hip/hip_bf16.h>

typedef __attribute__((ext_vector_type(16))) __bf16 v16bf;
typedef __attribute__((ext_vector_type(8)))  __bf16 v8bf;
typedef __attribute__((ext_vector_type(4)))  __bf16 v4bf;
typedef __attribute__((ext_vector_type(8)))  float  v8f;
typedef __attribute__((ext_vector_type(2)))  int    v2i;

#define BN_EPS 1e-3f
#define MAX_KPAD 1728            // 64 * 27
#define LDS_ROW  (MAX_KPAD + 16) // +16 bf16 (32B) row pad to stagger LDS banks

// ---- optional CDNA5 async global->LDS path (guarded; falls back cleanly) ----
#if defined(__has_builtin)
#if __has_builtin(__builtin_amdgcn_global_load_async_to_lds_b64)
#define HAVE_ASYNC_LDS 1
#endif
#if __has_builtin(__builtin_amdgcn_s_wait_asynccnt)
#define WAIT_ASYNCCNT0() __builtin_amdgcn_s_wait_asynccnt(0)
#endif
#endif
#ifndef WAIT_ASYNCCNT0
#define WAIT_ASYNCCNT0() asm volatile("s_wait_asynccnt 0x0" ::: "memory")
#endif

typedef __attribute__((address_space(1))) v2i* gv2i_ptr;
typedef __attribute__((address_space(3))) v2i* lv2i_ptr;

// ---------------------------------------------------------------------------
// Scatter voxels into dense bf16 NHWC grid ([site][C]) + active mask.
// ---------------------------------------------------------------------------
__global__ void vox_scatter_kernel(const float* __restrict__ feat,
                                   const int* __restrict__ lin,
                                   __bf16* __restrict__ act,
                                   unsigned char* __restrict__ mask,
                                   int nvox) {
  int t = blockIdx.x * blockDim.x + threadIdx.x;
  if (t >= nvox) return;
  int idx = lin[t];
  mask[idx] = 1;
  v4bf v;
#pragma unroll
  for (int c = 0; c < 4; ++c) v[c] = (__bf16)feat[t * 4 + c];
  *(v4bf*)(act + (size_t)idx * 4) = v;
}

// ---------------------------------------------------------------------------
// Pack weights (x folded BN scale) into the A-fragment lane layout of
// v_wmma_f32_16x16x32_bf16 (16-bit A 16x32):
//   contiguous k order (own choice, B reads match): element j holds K = ((lane&16)?16:0)+j,
//   M = lane&15.  GEMM K-order is k = tap*Cin + ci (NHWC gather order);
//   source Wt is OIDHW, i.e. Wt[co*K + ci*Kvol + tap].
// ---------------------------------------------------------------------------
__global__ void prep_weights_kernel(const float* __restrict__ Wt,
                                    const float* __restrict__ gamma,
                                    const float* __restrict__ var,
                                    __bf16* __restrict__ wfrag,
                                    int Cout, int K, int Kvol, int lcin,
                                    int nChunks, int nGroups) {
  int tid = blockIdx.x * blockDim.x + threadIdx.x;
  int total = nGroups * nChunks * 512;
  if (tid >= total) return;
  int j    = tid & 15;
  int lane = (tid >> 4) & 31;
  int rest = tid >> 9;
  int c    = rest % nChunks;
  int g    = rest / nChunks;
  int k  = c * 32 + ((lane & 16) ? 16 : 0) + j;
  int m  = lane & 15;
  int co = g * 16 + m;
  float v = 0.f;
  if (k < K && co < Cout) {
    int ci  = k & ((1 << lcin) - 1);   // Cin is a power of two
    int tap = k >> lcin;
    float sc = gamma[co] * rsqrtf(var[co] + BN_EPS); // fold BN scale into W
    v = Wt[(size_t)co * K + ci * Kvol + tap] * sc;
  }
  wfrag[tid] = (__bf16)v;
}

// ---------------------------------------------------------------------------
// Regular sparse-conv active-site propagation (OR over receptive field).
// ---------------------------------------------------------------------------
__global__ void mask_dilate_kernel(const unsigned char* __restrict__ mi,
                                   unsigned char* __restrict__ mo,
                                   int Din, int Hin, int Win,
                                   int Dout, int Hout, int Wout,
                                   int kd, int kh, int kw,
                                   int sd, int sh, int sw,
                                   int pd, int ph, int pw) {
  int s = blockIdx.x * blockDim.x + threadIdx.x;
  int ns = Dout * Hout * Wout;
  if (s >= ns) return;
  int ow = s % Wout; int t = s / Wout; int oh = t % Hout; int od = t / Hout;
  unsigned char any = 0;
  for (int a = 0; a < kd; ++a) {
    int id = od * sd - pd + a; if (id < 0 || id >= Din) continue;
    for (int b = 0; b < kh; ++b) {
      int ih = oh * sh - ph + b; if (ih < 0 || ih >= Hin) continue;
      for (int cc = 0; cc < kw; ++cc) {
        int iw = ow * sw - pw + cc; if (iw < 0 || iw >= Win) continue;
        any |= mi[(id * Hin + ih) * Win + iw];
      }
    }
  }
  mo[s] = any ? 1 : 0;
}

// ---------------------------------------------------------------------------
// Implicit-GEMM conv + folded BN + ReLU + mask. bf16 WMMA, f32 accumulate.
// Block = 128 threads (4 waves) = one 16-site N tile; waves split M groups.
// NHWC activations: interior tiles use async global->LDS B64 copies; border
// tiles use masked 8B loads with zero fill.
// ---------------------------------------------------------------------------
__global__ __launch_bounds__(128)
void conv_wmma_kernel(const __bf16* __restrict__ xin,
                      void* __restrict__ out, int out_f32,
                      const __bf16* __restrict__ wfrag,
                      const float* __restrict__ gamma,
                      const float* __restrict__ beta,
                      const float* __restrict__ mean,
                      const float* __restrict__ var,
                      const unsigned char* __restrict__ mask_out,
                      int Cin, int Cout, int lc4 /* log2(Cin/4) */,
                      int K, int Kpad, int nChunks, int nGroups,
                      int Din, int Hin, int Win,
                      int Dout, int Hout, int Wout,
                      int kd, int kh, int kw,
                      int sd, int sh, int sw,
                      int pd, int ph, int pw) {
  __shared__ alignas(64) __bf16 xlds[16 * LDS_ROW];
  __shared__ int sod[16], soh[16], sow[16], svld[16], sint[16];

  const int tid  = threadIdx.x;
  const int Ns   = Dout * Hout * Wout;
  const int tile = blockIdx.x;
  const int krow = Kpad + 16; // padded LDS row stride (elements, %16 == 0)

  // ---- decode the 16 output-site coordinates once (the only divisions)
  if (tid < 16) {
    int s = tile * 16 + tid;
    int valid = (s < Ns);
    int ow = s % Wout; int t2 = s / Wout; int oh = t2 % Hout; int od = t2 / Hout;
    int bd = od * sd - pd, bh = oh * sh - ph, bw = ow * sw - pw;
    sod[tid] = bd; soh[tid] = bh; sow[tid] = bw;
    svld[tid] = valid;
    sint[tid] = valid && bd >= 0 && (bd + kd) <= Din &&
                bh >= 0 && (bh + kh) <= Hin &&
                bw >= 0 && (bw + kw) <= Win;
  }
  __syncthreads();

  int interior = 1;
#pragma unroll
  for (int n = 0; n < 16; ++n) interior &= sint[n];

  // ---- gather: X patch -> LDS rows [n][k], k = tap*Cin + ci, 8B granules
  const int chunks4 = 1 << lc4;          // Cin/4
  const int work    = 16 << lc4;         // 16 sites * chunks4
#if defined(HAVE_ASYNC_LDS)
  if (interior) {
    // fast path: every tap of every site in bounds -> pure async copies
    for (int td = 0; td < kd; ++td)
      for (int th = 0; th < kh; ++th)
        for (int tw = 0; tw < kw; ++tw) {
          const int tap = (td * kh + th) * kw + tw;
          for (int w = tid; w < work; w += 128) {
            int n  = w >> lc4;
            int cc = (w & (chunks4 - 1)) << 2;
            size_t lin = (size_t)((sod[n] + td) * Hin + (soh[n] + th)) * Win +
                         (sow[n] + tw);
            __builtin_amdgcn_global_load_async_to_lds_b64(
                (gv2i_ptr)(xin + lin * Cin + cc),
                (lv2i_ptr)(&xlds[n * krow + tap * Cin + cc]),
                0, 0);
          }
        }
    WAIT_ASYNCCNT0();
  } else
#endif
  {
    for (int td = 0; td < kd; ++td)
      for (int th = 0; th < kh; ++th)
        for (int tw = 0; tw < kw; ++tw) {
          const int tap = (td * kh + th) * kw + tw;
          for (int w = tid; w < work; w += 128) {
            int n  = w >> lc4;
            int cc = (w & (chunks4 - 1)) << 2;
            int id = sod[n] + td;
            int ih = soh[n] + th;
            int iw = sow[n] + tw;
            unsigned long long val = 0ull;
            if (svld[n] && id >= 0 && id < Din && ih >= 0 && ih < Hin &&
                iw >= 0 && iw < Win) {
              size_t lin = (size_t)(id * Hin + ih) * Win + iw;
              val = *(const unsigned long long*)(xin + lin * Cin + cc);
            }
            *(unsigned long long*)(&xlds[n * krow + tap * Cin + cc]) = val;
          }
        }
  }
  // zero-fill K..Kpad tail (< 32 elements per row)
  for (int n = 0; n < 16; ++n)
    for (int k = K + tid; k < Kpad; k += 128)
      xlds[n * krow + k] = (__bf16)0.f;
  __syncthreads();

  // ---- WMMA main loop
  const int lane = tid & 31;
  const int wave = tid >> 5;
  const int nrow = lane & 15;
  const int hi   = (lane & 16) ? 1 : 0;
  const v16bf* bbase = reinterpret_cast<const v16bf*>(xlds);

  for (int g = wave; g < nGroups; g += 4) {
    v8f acc = {0.f, 0.f, 0.f, 0.f, 0.f, 0.f, 0.f, 0.f};
    const v16bf* abase =
        reinterpret_cast<const v16bf*>(wfrag) + (size_t)g * nChunks * 32;
    for (int c = 0; c < nChunks; ++c) {
      if (c + 1 < nChunks)  // prefetch next A chunk -> global_prefetch_b8
        __builtin_prefetch((const void*)(abase + (c + 1) * 32 + lane), 0, 0);
      v16bf a = abase[c * 32 + lane];                         // 32B global
      v16bf b = bbase[(nrow * krow + c * 32 + hi * 16) >> 4]; // 32B LDS
      acc = __builtin_amdgcn_wmma_f32_16x16x32_bf16(
          false, a, false, b, (short)0, acc, false, false);
    }
    // ---- epilogue: +shift (scale folded into W), ReLU, mask, packed store
    int site = tile * 16 + nrow;
    if (site < Ns) {
      float mk = (float)mask_out[site];
      int cobase = g * 16 + hi * 8;
      if (out_f32) {
        float* fo = (float*)out;
#pragma unroll
        for (int r = 0; r < 8; ++r) {
          int co = cobase + r;
          float sc  = gamma[co] * rsqrtf(var[co] + BN_EPS);
          float shf = beta[co] - mean[co] * sc;
          fo[(size_t)co * Ns + site] = fmaxf(acc[r] + shf, 0.f) * mk; // NCDHW
        }
      } else {
        v8bf pk;
#pragma unroll
        for (int r = 0; r < 8; ++r) {
          int co = cobase + r;
          float sc  = gamma[co] * rsqrtf(var[co] + BN_EPS);
          float shf = beta[co] - mean[co] * sc;
          pk[r] = (__bf16)(fmaxf(acc[r] + shf, 0.f) * mk);
        }
        *(v8bf*)((__bf16*)out + (size_t)site * Cout + cobase) = pk; // NHWC
      }
    }
  }
}

// ---------------------------------------------------------------------------
// Host orchestration
// ---------------------------------------------------------------------------
extern "C" void kernel_launch(void* const* d_in, const int* in_sizes, int n_in,
                              void* d_out, int out_size, void* d_ws, size_t ws_size,
                              hipStream_t stream) {
  (void)in_sizes; (void)n_in; (void)out_size; (void)ws_size;
  const float* feat = (const float*)d_in[0];
  const int*   lin  = (const int*)d_in[1];

  // ---- layer configs (mirror of CFG in the reference)
  const int CI[12]   = {4,16,16,32,32,32,64,64,64,64,64,64};
  const int CO[12]   = {16,16,32,32,32,64,64,64,64,64,64,128};
  const int LC4[12]  = {0,2,2,3,3,3,4,4,4,4,4,4};  // log2(Cin/4)
  const int LCIN[12] = {2,4,4,5,5,5,6,6,6,6,6,6};  // log2(Cin)
  const int KD[12]   = {3,3,3,3,3,3,3,3,3,3,3,3};
  const int KH[12]   = {3,3,3,3,3,3,3,3,3,3,3,1};
  const int KW[12]   = {3,3,3,3,3,3,3,3,3,3,3,1};
  const int SD[12]   = {1,1,2,1,1,2,1,1,2,1,1,2};
  const int SH[12]   = {1,1,2,1,1,2,1,1,2,1,1,1};
  const int SW[12]   = {1,1,2,1,1,2,1,1,2,1,1,1};
  const int PD[12]   = {1,1,1,1,1,1,1,1,0,1,1,0};
  const int PH[12]   = {1,1,1,1,1,1,1,1,1,1,1,0};
  const int PW[12]   = {1,1,1,1,1,1,1,1,1,1,1,0};
  const int SUBM[12] = {1,1,0,1,1,0,1,1,0,1,1,0};
  const int INL[12]  = {0,0,0,1,1,1,2,2,2,3,3,3};
  const int OUTL[12] = {0,0,1,1,1,2,2,2,3,3,3,4};
  const int LDim[5]  = {41,21,11,5,2};
  const int LHgt[5]  = {320,160,80,40,40};
  const int LWid[5]  = {320,160,80,40,40};

  // ---- carve workspace
  char* base = (char*)d_ws;
  size_t off = 0;
  auto carve = [&](size_t bytes) -> void* {
    void* p = (void*)(base + off);
    off += (bytes + 255) & ~(size_t)255;
    return p;
  };
  const size_t ns0 = (size_t)41 * 320 * 320;
  __bf16* actA = (__bf16*)carve(16 * ns0 * sizeof(__bf16));
  __bf16* actB = (__bf16*)carve(16 * ns0 * sizeof(__bf16));
  unsigned char* masks[5];
  for (int l = 0; l < 5; ++l)
    masks[l] = (unsigned char*)carve((size_t)LDim[l] * LHgt[l] * LWid[l]);
  __bf16* wfrag[12];
  int Kv[12], Kk[12], Kp[12], NC[12], NG[12];
  for (int l = 0; l < 12; ++l) {
    Kv[l] = KD[l] * KH[l] * KW[l];
    Kk[l] = CI[l] * Kv[l];
    NC[l] = (Kk[l] + 31) / 32;
    Kp[l] = NC[l] * 32;
    NG[l] = CO[l] / 16;
    wfrag[l] = (__bf16*)carve((size_t)NG[l] * NC[l] * 512 * sizeof(__bf16));
  }

  // ---- scatter voxels (NHWC)
  (void)hipMemsetAsync(actA, 0, 4 * ns0 * sizeof(__bf16), stream);
  (void)hipMemsetAsync(masks[0], 0, ns0, stream);
  {
    int nvox = 60000;
    vox_scatter_kernel<<<(nvox + 255) / 256, 256, 0, stream>>>(
        feat, lin, actA, masks[0], nvox);
  }

  // ---- layers
  const __bf16* cur = actA;
  __bf16* nxt = actB;
  for (int l = 0; l < 12; ++l) {
    const float* Wt    = (const float*)d_in[2 + l * 5 + 0];
    const float* gamma = (const float*)d_in[2 + l * 5 + 1];
    const float* beta  = (const float*)d_in[2 + l * 5 + 2];
    const float* mean  = (const float*)d_in[2 + l * 5 + 3];
    const float* var   = (const float*)d_in[2 + l * 5 + 4];

    int il = INL[l], ol = OUTL[l];
    int Din = LDim[il], Hin = LHgt[il], Win = LWid[il];
    int Dout = LDim[ol], Hout = LHgt[ol], Wout = LWid[ol];
    int Ns = Dout * Hout * Wout;

    { // pack weights (BN scale folded, K remapped to NHWC order)
      int total = NG[l] * NC[l] * 512;
      prep_weights_kernel<<<(total + 255) / 256, 256, 0, stream>>>(
          Wt, gamma, var, wfrag[l], CO[l], Kk[l], Kv[l], LCIN[l], NC[l], NG[l]);
    }
    if (!SUBM[l]) { // regular sparse conv: dilate mask
      mask_dilate_kernel<<<(Ns + 255) / 256, 256, 0, stream>>>(
          masks[il], masks[ol], Din, Hin, Win, Dout, Hout, Wout,
          KD[l], KH[l], KW[l], SD[l], SH[l], SW[l], PD[l], PH[l], PW[l]);
    }
    int out_f32 = (l == 11);
    void* outp = out_f32 ? d_out : (void*)nxt;
    conv_wmma_kernel<<<Ns / 16, 128, 0, stream>>>(
        cur, outp, out_f32, wfrag[l], gamma, beta, mean, var, masks[ol],
        CI[l], CO[l], LC4[l], Kk[l], Kp[l], NC[l], NG[l],
        Din, Hin, Win, Dout, Hout, Wout,
        KD[l], KH[l], KW[l], SD[l], SH[l], SW[l], PD[l], PH[l], PW[l]);
    // ping-pong
    const __bf16* t = cur;
    cur = nxt;
    nxt = (__bf16*)t;
  }
}
